// RPN_80427557585243
// MI455X (gfx1250) — compile-verified, matrix operand order checked by reading, not run on hardware
//
#include <hip/hip_runtime.h>

typedef __bf16 bf16;
typedef __attribute__((ext_vector_type(16))) __bf16 v16bf;
typedef __attribute__((ext_vector_type(8)))  __bf16 v8bf;
typedef __attribute__((ext_vector_type(8)))  float  v8f;

#define NPIX     4096      // 64*64
#define CIN      256
#define NANCH    36864     // 4096*9
#define PRE_NMS  3000
#define POST_NMS 256
#define CAND_CAP 4096

// ---------------------------------------------------------------------------
// Layout conversion: NCHW f32 -> NHWC bf16
// ---------------------------------------------------------------------------
__global__ void to_nhwc_kernel(const float* __restrict__ src, bf16* __restrict__ dst) {
  int idx = blockIdx.x * blockDim.x + threadIdx.x;   // 4096*256
  if (idx >= NPIX * CIN) return;
  int c = idx & 255;
  int p = idx >> 8;
  dst[idx] = (bf16)src[c * NPIX + p];
}

// ---------------------------------------------------------------------------
// Pack OIHW f32 weights into per-lane WMMA B-fragment bf16 layout:
//   dst[((t*8+q)*nchunks + ntile)*512 + lane*16 + i]
//     = W[n = ntile*16 + (lane&15)][c = q*32 + klocal][tap t]
//   klocal = (i&7) + 8*(lane>>4) + 16*(i>>3)   (16-bit B fragment, wave32)
// ---------------------------------------------------------------------------
__global__ void pack_weights_kernel(const float* __restrict__ w, bf16* __restrict__ dst,
                                    int cout_valid, int nchunks) {
  int idx = blockIdx.x * blockDim.x + threadIdx.x;
  int total = 9 * 8 * nchunks * 512;
  if (idx >= total) return;
  int i    = idx & 15;
  int lane = (idx >> 4) & 31;
  int rest = idx >> 9;
  int ntile = rest % nchunks;  rest /= nchunks;
  int q     = rest & 7;
  int t     = rest >> 3;
  int n      = ntile * 16 + (lane & 15);
  int klocal = (i & 7) + 8 * (lane >> 4) + 16 * (i >> 3);
  int c      = q * 32 + klocal;
  float v = (n < cout_valid) ? w[(n * CIN + c) * 9 + t] : 0.0f;
  dst[idx] = (bf16)v;
}

// ---------------------------------------------------------------------------
// Tower conv (3x3 SAME, 256->256) via WMMA bf16, LDS-staged activations.
// Block = 256 threads = 8 waves = one feature row x 128 couts.
//   blockIdx.x = h*2 + jhalf ; wave handles ntile = jhalf*8 + wave (16 couts)
//   and 4 M-subtiles (the whole 64-pixel row) -> 4 WMMA per B fragment.
// LDS tile: [3 halo rows][66 padded pixels][32 chans] bf16, restaged per
// 32-channel K-chunk; zero padding kills all boundary branches in the loop.
// ---------------------------------------------------------------------------
__global__ __launch_bounds__(256) void conv_row_wmma_kernel(
    const bf16*  __restrict__ xin,    // [4096][256] NHWC bf16
    const bf16*  __restrict__ wp,     // packed weights (nchunks = 16)
    const float* __restrict__ bias,   // [256]
    bf16*        __restrict__ obf)    // [4096][256] NHWC bf16
{
  __shared__ __align__(16) bf16 tile[3 * 66 * 32];   // 12672 B
  const int tid   = threadIdx.x;
  const int lane  = tid & 31;
  const int wave  = tid >> 5;            // 0..7
  const int h     = blockIdx.x >> 1;
  const int jhalf = blockIdx.x & 1;
  const int ntile = jhalf * 8 + wave;    // 0..15
  const int m     = lane & 15;
  const int half  = lane >> 4;

  v8f acc[4] = {{}, {}, {}, {}};

  for (int q = 0; q < 8; ++q) {
    __syncthreads();
    // stage 3 rows x 66 px x 32 ch (zero-padded halo), 16B chunks
    for (int ch = tid; ch < 792; ch += 256) {
      int row = ch / 264;
      int rem = ch - row * 264;
      int px  = rem >> 2;                // 0..65  (pixel -1..64)
      int cc  = (rem & 3) * 16;          // byte offset inside 64B pixel chunk
      int hh  = h + row - 1;
      int pg  = px - 1;
      uint4 v = make_uint4(0u, 0u, 0u, 0u);
      if ((unsigned)hh < 64u && (unsigned)pg < 64u)
        v = *(const uint4*)((const char*)xin + (size_t)(hh * 64 + pg) * 512 + q * 64 + cc);
      *(uint4*)((char*)tile + (size_t)(row * 66 + px) * 64 + cc) = v;
    }
    __syncthreads();

    const bf16* bpq = wp + (((q * 16 + ntile) << 9) + (lane << 4));
#pragma unroll
    for (int t = 0; t < 9; ++t) {
      const int dh1 = t / 3;             // halo row 0..2
      const int dw  = t % 3;             // tile px offset 0..2
      // B fragment: ((t*8+q)*16 + ntile)*512 -> bpq + t*65536
      v16bf b = *(const v16bf*)(bpq + (t << 16));
      const bf16* arow = tile + (dh1 * 66 + m + dw) * 32 + (half << 3);
#pragma unroll
      for (int s = 0; s < 4; ++s) {
        const bf16* ap = arow + s * (16 * 32);
        v8bf lo = *(const v8bf*)(ap);
        v8bf hi = *(const v8bf*)(ap + 16);
        v16bf a = __builtin_shufflevector(lo, hi,
                    0,1,2,3,4,5,6,7,8,9,10,11,12,13,14,15);
        acc[s] = __builtin_amdgcn_wmma_f32_16x16x32_bf16(
            false, a, false, b, (short)0, acc[s], false, false);
      }
    }
  }

  const int n  = (ntile << 4) + m;
  const float bv = bias[n];
#pragma unroll
  for (int s = 0; s < 4; ++s)
#pragma unroll
    for (int r = 0; r < 8; ++r) {
      const int pixel = h * 64 + s * 16 + r + (half << 3);
      obf[(pixel << 8) + n] = (bf16)(acc[s][r] + bv);
    }
}

// ---------------------------------------------------------------------------
// Head conv (3x3 SAME, 256 -> cout, cout padded to NCHUNKS*16), f32 out.
// One wave per (16-pixel x 16-cout) tile; NCHUNKS is compile-time (no idiv).
// ---------------------------------------------------------------------------
template<int NCHUNKS>
__global__ __launch_bounds__(128) void conv_head_kernel(
    const bf16*  __restrict__ xin,    // [4096][256] NHWC bf16
    const bf16*  __restrict__ wp,     // packed weights
    const float* __restrict__ bias,   // [cout_valid]
    float*       __restrict__ of32,   // [4096][cout_valid]
    int cout_valid)
{
  const int gwave = (blockIdx.x * 128 + threadIdx.x) >> 5;
  const int lane  = threadIdx.x & 31;
  const int ntile = gwave % NCHUNKS;
  const int mtile = gwave / NCHUNKS;
  if (mtile >= 256) return;
  const int h    = mtile >> 2;
  const int w0   = (mtile & 3) << 4;
  const int m    = lane & 15;
  const int half = lane >> 4;

  v8f acc = {};
  for (int t = 0; t < 9; ++t) {
    const int hh = h + (t / 3) - 1;
    const int ww = w0 + m + (t % 3) - 1;
    const bool valid = ((unsigned)hh < 64u) && ((unsigned)ww < 64u);
    const bf16* xp = xin + (((hh * 64 + ww) << 8) + (half << 3));
    const bf16* bp = wp + (((t * 8) * NCHUNKS + ntile) << 9) + (lane << 4);
#pragma unroll
    for (int q = 0; q < 8; ++q) {
      v8bf z = {};
      v8bf lo = valid ? *(const v8bf*)(xp + q * 32)      : z;
      v8bf hi = valid ? *(const v8bf*)(xp + q * 32 + 16) : z;
      v16bf a = __builtin_shufflevector(lo, hi,
                  0,1,2,3,4,5,6,7,8,9,10,11,12,13,14,15);
      v16bf b = *(const v16bf*)(bp + ((q * NCHUNKS) << 9));
      acc = __builtin_amdgcn_wmma_f32_16x16x32_bf16(
          false, a, false, b, (short)0, acc, false, false);
    }
  }
  const int n  = (ntile << 4) + m;
  const float bv = (n < cout_valid) ? bias[n] : 0.0f;
#pragma unroll
  for (int r = 0; r < 8; ++r) {
    const int pixel = h * 64 + w0 + r + (half << 3);
    if (n < cout_valid) of32[pixel * cout_valid + n] = acc[r] + bv;
  }
}

// ---------------------------------------------------------------------------
// Per-anchor softmax over 2 classes + objectness score
// ---------------------------------------------------------------------------
__global__ void softmax_kernel(const float* __restrict__ logits,   // [4096][18]
                               float* __restrict__ cls_out,        // [4096][18]
                               float* __restrict__ score) {        // [36864]
  int i = blockIdx.x * blockDim.x + threadIdx.x;
  if (i >= NANCH) return;
  int p = i / 9, a = i % 9;
  float l0 = logits[p * 18 + a * 2 + 0];
  float l1 = logits[p * 18 + a * 2 + 1];
  float mx = fmaxf(l0, l1);
  float e0 = __expf(l0 - mx), e1 = __expf(l1 - mx);
  float inv = 1.0f / (e0 + e1);
  cls_out[p * 18 + a * 2 + 0] = e0 * inv;
  cls_out[p * 18 + a * 2 + 1] = e1 * inv;
  score[i] = 1.0f - e0 * inv;
}

// ---------------------------------------------------------------------------
// Box decode + clip + min-size filter + sortable key
// ---------------------------------------------------------------------------
__global__ void decode_kernel(const float* __restrict__ anchors,
                              const float* __restrict__ loc,       // [4096][36]
                              float* __restrict__ score,           // in/out
                              float* __restrict__ boxes,           // [36864][4]
                              unsigned* __restrict__ keys) {
  int i = blockIdx.x * blockDim.x + threadIdx.x;
  if (i >= NANCH) return;
  int p = i / 9, a = i % 9;
  float ax1 = anchors[i*4+0], ay1 = anchors[i*4+1];
  float ax2 = anchors[i*4+2], ay2 = anchors[i*4+3];
  float aw = ax2 - ax1, ah = ay2 - ay1;
  float acx = ax1 + 0.5f * aw, acy = ay1 + 0.5f * ah;
  const float* L = loc + p * 36 + a * 4;
  float cx = L[0] * aw + acx, cy = L[1] * ah + acy;
  float w  = __expf(L[2]) * aw, hgt = __expf(L[3]) * ah;
  float x1 = fminf(fmaxf(cx - 0.5f * w,   0.f), 1024.f);
  float y1 = fminf(fmaxf(cy - 0.5f * hgt, 0.f), 1024.f);
  float x2 = fminf(fmaxf(cx + 0.5f * w,   0.f), 1024.f);
  float y2 = fminf(fmaxf(cy + 0.5f * hgt, 0.f), 1024.f);
  boxes[i*4+0]=x1; boxes[i*4+1]=y1; boxes[i*4+2]=x2; boxes[i*4+3]=y2;
  float sc = score[i];
  if (!((x2 - x1) >= 16.0f && (y2 - y1) >= 16.0f)) sc = -1.0f;
  score[i] = sc;
  unsigned b = __float_as_uint(sc);
  keys[i] = (b & 0x80000000u) ? ~b : (b | 0x80000000u);
}

// ---------------------------------------------------------------------------
// 32-pass bitwise radix select of the PRE_NMS-th largest key (one workgroup)
// ---------------------------------------------------------------------------
__global__ __launch_bounds__(1024) void select_kernel(const unsigned* __restrict__ keys,
                                                      unsigned* __restrict__ counters) {
  __shared__ int red[1024];
  const int tid = threadIdx.x;
  unsigned T = 0;
  for (int bit = 31; bit >= 0; --bit) {
    unsigned candv = T | (1u << bit);
    int c = 0;
    for (int i = tid; i < NANCH; i += 1024) c += (keys[i] >= candv) ? 1 : 0;
    red[tid] = c;
    __syncthreads();
    for (int s = 512; s > 0; s >>= 1) {
      if (tid < s) red[tid] += red[tid + s];
      __syncthreads();
    }
    if (red[0] >= PRE_NMS) T = candv;
    __syncthreads();
  }
  if (tid == 0) counters[0] = T;
}

// ---------------------------------------------------------------------------
// Compact all keys >= threshold into candidate list (u64: key<<32 | ~index)
// ---------------------------------------------------------------------------
__global__ void compact_kernel(const unsigned* __restrict__ keys,
                               const unsigned* __restrict__ counters,
                               unsigned long long* __restrict__ cand,
                               unsigned* __restrict__ counter) {
  int i = blockIdx.x * blockDim.x + threadIdx.x;
  if (i >= NANCH) return;
  unsigned k = keys[i];
  if (k >= counters[0]) {
    unsigned pos = atomicAdd(counter, 1u);
    if (pos < CAND_CAP)
      cand[pos] = ((unsigned long long)k << 32) | (unsigned)(~(unsigned)i);
  }
}

// ---------------------------------------------------------------------------
// Bitonic sort 4096 u64 keys descending (score desc, index asc), one WG, LDS
// ---------------------------------------------------------------------------
__global__ __launch_bounds__(1024) void sort_kernel(unsigned long long* __restrict__ cand) {
  __shared__ unsigned long long s[CAND_CAP];
  const int tid = threadIdx.x;
  for (int i = tid; i < CAND_CAP; i += 1024) s[i] = cand[i];
  __syncthreads();
  for (int k = 2; k <= CAND_CAP; k <<= 1) {
    for (int j = k >> 1; j > 0; j >>= 1) {
      for (int i = tid; i < CAND_CAP; i += 1024) {
        int ixj = i ^ j;
        if (ixj > i) {
          unsigned long long a = s[i], b = s[ixj];
          bool desc = ((i & k) == 0);
          if (desc ? (a < b) : (a > b)) { s[i] = b; s[ixj] = a; }
        }
      }
      __syncthreads();
    }
  }
  for (int i = tid; i < CAND_CAP; i += 1024) cand[i] = s[i];
}

// ---------------------------------------------------------------------------
// Greedy NMS over top-3000 (sorted): sequential over i, parallel over j
// ---------------------------------------------------------------------------
__global__ __launch_bounds__(1024) void nms_kernel(
    const unsigned long long* __restrict__ cand,
    const float* __restrict__ boxes,
    unsigned char* __restrict__ sup_g) {
  __shared__ float bx1[PRE_NMS], by1[PRE_NMS], bx2[PRE_NMS], by2[PRE_NMS], ar[PRE_NMS];
  __shared__ unsigned char sup[PRE_NMS];
  const int tid = threadIdx.x;
  for (int i = tid; i < PRE_NMS; i += 1024) {
    unsigned idx = ~((unsigned)(cand[i] & 0xFFFFFFFFull));
    if (idx >= (unsigned)NANCH) idx = 0u;
    float x1 = boxes[idx*4+0], y1 = boxes[idx*4+1];
    float x2 = boxes[idx*4+2], y2 = boxes[idx*4+3];
    bx1[i]=x1; by1[i]=y1; bx2[i]=x2; by2[i]=y2;
    ar[i] = (x2 - x1) * (y2 - y1);
    sup[i] = 0;
  }
  __syncthreads();
  for (int i = 0; i < PRE_NMS; ++i) {
    __syncthreads();
    if (sup[i]) continue;                 // uniform across the block
    float x1 = bx1[i], y1 = by1[i], x2 = bx2[i], y2 = by2[i], ai = ar[i];
    for (int j = i + 1 + tid; j < PRE_NMS; j += 1024) {
      float ix1 = fmaxf(x1, bx1[j]), iy1 = fmaxf(y1, by1[j]);
      float ix2 = fminf(x2, bx2[j]), iy2 = fminf(y2, by2[j]);
      float inter = fmaxf(ix2 - ix1, 0.f) * fmaxf(iy2 - iy1, 0.f);
      float iou = inter / (ai + ar[j] - inter + 1e-9f);
      if (iou > 0.7f) sup[j] = 1;
    }
  }
  __syncthreads();
  for (int i = tid; i < PRE_NMS; i += 1024) sup_g[i] = sup[i];
}

// ---------------------------------------------------------------------------
// Pick POST_NMS rois: unsuppressed first (sorted order), then fill with -1s
// ---------------------------------------------------------------------------
__global__ void roi_select_kernel(const unsigned long long* __restrict__ cand,
                                  const unsigned char* __restrict__ sup,
                                  const float* __restrict__ boxes,
                                  const float* __restrict__ score,
                                  float* __restrict__ roi,
                                  float* __restrict__ roi_score,
                                  float* __restrict__ roi_label) {
  if (threadIdx.x != 0 || blockIdx.x != 0) return;
  int cnt = 0;
  for (int pass = 0; pass < 2 && cnt < POST_NMS; ++pass) {
    for (int i = 0; i < PRE_NMS && cnt < POST_NMS; ++i) {
      bool s = sup[i] != 0;
      if ((pass == 0 && s) || (pass == 1 && !s)) continue;
      unsigned idx = ~((unsigned)(cand[i] & 0xFFFFFFFFull));
      if (idx >= (unsigned)NANCH) idx = 0u;
      roi[cnt*4+0] = boxes[idx*4+0];
      roi[cnt*4+1] = boxes[idx*4+1];
      roi[cnt*4+2] = boxes[idx*4+2];
      roi[cnt*4+3] = boxes[idx*4+3];
      roi_score[cnt] = s ? -1.0f : score[idx];
      roi_label[cnt] = 1.0f;
      ++cnt;
    }
  }
}

// ---------------------------------------------------------------------------
// ROI max pool 7x7: one block per ROI, one thread per channel
// ---------------------------------------------------------------------------
__global__ __launch_bounds__(256) void roi_pool_kernel(const float* __restrict__ f, // NCHW
                                                       const float* __restrict__ roi,
                                                       float* __restrict__ out) {
  const int r = blockIdx.x;
  const int c = threadIdx.x;
  const float sc = 0.0625f;                 // 1/16
  int x1 = (int)rintf(roi[r*4+0] * sc);
  int y1 = (int)rintf(roi[r*4+1] * sc);
  int x2 = (int)rintf(roi[r*4+2] * sc);
  int y2 = (int)rintf(roi[r*4+3] * sc);
  float rw = fmaxf((float)(x2 - x1 + 1), 1.0f);
  float rh = fmaxf((float)(y2 - y1 + 1), 1.0f);
  const float* fc = f + c * NPIX;
  for (int ph = 0; ph < 7; ++ph) {
    int hs = min(max((int)floorf(ph * rh * (1.0f/7.0f)) + y1, 0), 64);
    int he = min(max((int)ceilf((ph + 1) * rh * (1.0f/7.0f)) + y1, 0), 64);
    for (int pw = 0; pw < 7; ++pw) {
      int wsx = min(max((int)floorf(pw * rw * (1.0f/7.0f)) + x1, 0), 64);
      int wex = min(max((int)ceilf((pw + 1) * rw * (1.0f/7.0f)) + x1, 0), 64);
      float m = -1e30f;
      for (int hh = hs; hh < he; ++hh)
        for (int ww = wsx; ww < wex; ++ww)
          m = fmaxf(m, fc[hh * 64 + ww]);
      out[r * 12544 + c * 49 + ph * 7 + pw] = (m < -5e29f) ? 0.0f : m;
    }
  }
}

// ---------------------------------------------------------------------------
// Zero-init scratch that must start clean every call
// ---------------------------------------------------------------------------
__global__ void init_kernel(unsigned long long* __restrict__ cand,
                            unsigned* __restrict__ counters,
                            unsigned char* __restrict__ sup) {
  int i = blockIdx.x * blockDim.x + threadIdx.x;
  if (i < CAND_CAP) cand[i] = 0ull;
  if (i < 64) counters[i] = 0u;
  if (i < PRE_NMS) sup[i] = 0;
}

// ---------------------------------------------------------------------------
extern "C" void kernel_launch(void* const* d_in, const int* in_sizes, int n_in,
                              void* d_out, int out_size, void* d_ws, size_t ws_size,
                              hipStream_t stream) {
  const float* features = (const float*)d_in[0];
  const float* anchors  = (const float*)d_in[1];
  const float* cls_w    = (const float*)d_in[2];
  const float* cls_b    = (const float*)d_in[3];
  const float* cls_wf   = (const float*)d_in[4];
  const float* cls_bf   = (const float*)d_in[5];
  const float* loc_w    = (const float*)d_in[6];
  const float* loc_b    = (const float*)d_in[7];
  const float* loc_wf   = (const float*)d_in[8];
  const float* loc_bf   = (const float*)d_in[9];
  float* out = (float*)d_out;

  // d_out regions (flat concatenation in return order, all f32)
  const size_t O_POOL = 0;
  const size_t O_CLS  = 3211264;   // + 256*256*49
  const size_t O_LOC  = 3284992;   // + 73728
  const size_t O_ROI  = 3432448;   // + 147456
  const size_t O_SC   = 3433472;   // + 1024
  const size_t O_LB   = 3433728;   // + 256

  // workspace carve-out
  char* ws = (char*)d_ws;
  size_t off = 0;
  auto carve = [&](size_t bytes) {
    char* p = ws + off;
    off = (off + bytes + 255) & ~(size_t)255;
    return p;
  };
  const size_t LPACK = 9 * 8 * 16 * 512;           // packed elems / tower layer
  bf16*  xnhwc   = (bf16*)carve(NPIX * CIN * 2);
  bf16*  act0    = (bf16*)carve(NPIX * CIN * 2);
  bf16*  act1    = (bf16*)carve(NPIX * CIN * 2);
  bf16*  wp_cls  = (bf16*)carve(5 * LPACK * 2);
  bf16*  wp_loc  = (bf16*)carve(5 * LPACK * 2);
  bf16*  wp_clsf = (bf16*)carve((size_t)9 * 8 * 2 * 512 * 2);
  bf16*  wp_locf = (bf16*)carve((size_t)9 * 8 * 3 * 512 * 2);
  float* cls_log = (float*)carve((size_t)NPIX * 18 * 4);
  float* score   = (float*)carve((size_t)NANCH * 4);
  float* boxes   = (float*)carve((size_t)NANCH * 4 * 4);
  unsigned* keys = (unsigned*)carve((size_t)NANCH * 4);
  unsigned* counters = (unsigned*)carve(256);
  unsigned long long* cand = (unsigned long long*)carve((size_t)CAND_CAP * 8);
  unsigned char* sup = (unsigned char*)carve(PRE_NMS + 64);

  // 0) zero scratch that must be clean
  init_kernel<<<16, 256, 0, stream>>>(cand, counters, sup);

  // 1) features -> NHWC bf16
  to_nhwc_kernel<<<(NPIX * CIN) / 256, 256, 0, stream>>>(features, xnhwc);

  // 2) pack weights into WMMA fragment layout
  for (int l = 0; l < 5; ++l) {
    pack_weights_kernel<<<144 * 16, 256, 0, stream>>>(
        cls_w + (size_t)l * CIN * CIN * 9, wp_cls + (size_t)l * LPACK, 256, 16);
    pack_weights_kernel<<<144 * 16, 256, 0, stream>>>(
        loc_w + (size_t)l * CIN * CIN * 9, wp_loc + (size_t)l * LPACK, 256, 16);
  }
  pack_weights_kernel<<<144 * 2, 256, 0, stream>>>(cls_wf, wp_clsf, 18, 2);
  pack_weights_kernel<<<144 * 3, 256, 0, stream>>>(loc_wf, wp_locf, 36, 3);

  // 3) cls tower: 5x (3x3 conv 256->256) + head conv 256->18
  conv_row_wmma_kernel<<<128, 256, 0, stream>>>(xnhwc, wp_cls + 0*LPACK, cls_b + 0*256, act0);
  conv_row_wmma_kernel<<<128, 256, 0, stream>>>(act0,  wp_cls + 1*LPACK, cls_b + 1*256, act1);
  conv_row_wmma_kernel<<<128, 256, 0, stream>>>(act1,  wp_cls + 2*LPACK, cls_b + 2*256, act0);
  conv_row_wmma_kernel<<<128, 256, 0, stream>>>(act0,  wp_cls + 3*LPACK, cls_b + 3*256, act1);
  conv_row_wmma_kernel<<<128, 256, 0, stream>>>(act1,  wp_cls + 4*LPACK, cls_b + 4*256, act0);
  conv_head_kernel<2><<<128, 128, 0, stream>>>(act0, wp_clsf, cls_bf, cls_log, 18);

  // 4) loc tower: 5x (3x3 conv 256->256) + head conv 256->36 (into d_out)
  conv_row_wmma_kernel<<<128, 256, 0, stream>>>(xnhwc, wp_loc + 0*LPACK, loc_b + 0*256, act0);
  conv_row_wmma_kernel<<<128, 256, 0, stream>>>(act0,  wp_loc + 1*LPACK, loc_b + 1*256, act1);
  conv_row_wmma_kernel<<<128, 256, 0, stream>>>(act1,  wp_loc + 2*LPACK, loc_b + 2*256, act0);
  conv_row_wmma_kernel<<<128, 256, 0, stream>>>(act0,  wp_loc + 3*LPACK, loc_b + 3*256, act1);
  conv_row_wmma_kernel<<<128, 256, 0, stream>>>(act1,  wp_loc + 4*LPACK, loc_b + 4*256, act0);
  conv_head_kernel<3><<<192, 128, 0, stream>>>(act0, wp_locf, loc_bf, out + O_LOC, 36);

  // 5) softmax + objectness (also writes cls_softmax output)
  softmax_kernel<<<144, 256, 0, stream>>>(cls_log, out + O_CLS, score);

  // 6) decode boxes, clip, min-size filter, sortable keys
  decode_kernel<<<144, 256, 0, stream>>>(anchors, out + O_LOC, score, boxes, keys);

  // 7) top-3000 via radix-select threshold + compaction + bitonic sort
  select_kernel<<<1, 1024, 0, stream>>>(keys, counters);
  compact_kernel<<<144, 256, 0, stream>>>(keys, counters, cand, counters + 1);
  sort_kernel<<<1, 1024, 0, stream>>>(cand);

  // 8) NMS
  nms_kernel<<<1, 1024, 0, stream>>>(cand, boxes, sup);

  // 9) pick 256 ROIs (writes roi / roi_score / roi_label)
  roi_select_kernel<<<1, 32, 0, stream>>>(cand, sup, boxes, score,
                                          out + O_ROI, out + O_SC, out + O_LB);

  // 10) ROI max pool on the original fp32 NCHW features
  roi_pool_kernel<<<POST_NMS, 256, 0, stream>>>(features, out + O_ROI, out + O_POOL);
}
